// Continue_Recovery_79345225826756
// MI455X (gfx1250) — matrix operand order repaired and checked by reading, not compiled, not run on hardware
//
#include <hip/hip_runtime.h>
#include <cstdint>

typedef __attribute__((ext_vector_type(16))) _Float16 v16h;
typedef __attribute__((ext_vector_type(8)))  float    v8f;

#define EPS_BN 1e-5f

// ---------------------------------------------------------------------------
// WMMA helpers (v_wmma_f32_16x16x32_f16)
// A frag: lane m=tid&15, hi=tid>>4 holds row m, K = hi*16 + e (contiguous)
// B frag: lane holds col n=tid&15, K = (tid>>4)*16 + e  (prepacked contiguously)
// D frag: element r -> row r + 8*(tid>>4), col = 16*j + (tid&15)
// ---------------------------------------------------------------------------
__device__ __forceinline__ v8f wmma_f16(v16h a, v16h b, v8f c) {
    return __builtin_amdgcn_wmma_f32_16x16x32_f16(false, a, false, b,
                                                  (short)0, c, false, false);
}

__device__ __forceinline__ v8f zero_acc() {
    v8f z;
#pragma unroll
    for (int i = 0; i < 8; ++i) z[i] = 0.f;
    return z;
}

__device__ __forceinline__ v16h zero_frag() {
    v16h z;
    ((float4*)&z)[0] = make_float4(0.f, 0.f, 0.f, 0.f);
    ((float4*)&z)[1] = make_float4(0.f, 0.f, 0.f, 0.f);
    return z;
}

// row points at the 32 K-values of this chunk for A-row m (contiguous f16)
__device__ __forceinline__ v16h load_a_frag(const _Float16* row, int hi) {
    v16h a;
    ((float4*)&a)[0] = *(const float4*)(row + hi * 16);
    ((float4*)&a)[1] = *(const float4*)(row + hi * 16 + 8);
    return a;
}

// pk points at 16 contiguous prepacked halves for this lane
__device__ __forceinline__ v16h load_b_frag(const _Float16* pk) {
    v16h b;
    ((float4*)&b)[0] = *(const float4*)(pk);
    ((float4*)&b)[1] = *(const float4*)(pk + 8);
    return b;
}

// ---------------------------------------------------------------------------
// Weight packing: fold BN into f16 weights, pre-swizzle into WMMA B layout:
//   packed[((chunk*NT + j)*32 + lane)*16 + e] = W[chunk*32 + (lane>>4)*16 + e][j*16 + (lane&15)]
// ---------------------------------------------------------------------------
__global__ void pack_weights(
    const float* reduce_w, const float* reduce_b,
    const float* bn1_g, const float* bn1_b, const float* bn1_m, const float* bn1_v,
    const float* off_w,
    const float* mlp_w, const float* mlp_b,
    const float* mbn_g, const float* mbn_b, const float* mbn_m, const float* mbn_v,
    const float* out_w1, const float* out_b1,
    const float* bn2_g, const float* bn2_b, const float* bn2_m, const float* bn2_v,
    _Float16* w1p, _Float16* rwp, _Float16* owp, _Float16* mwp,
    float* rbias, float* mbias, float* bias1)
{
    const int N_W1 = 36 * 4 * 32 * 16;   // 73728  (out conv: K=1152, N=64)
    const int N_RW = 12 * 2 * 32 * 16;   // 12288  (reduce: K=384, N=32)
    const int N_OW = 9 * 32 * 16;        // 4608   (offset: K=288, N=16 pad)
    const int N_MW = 3 * 2 * 32 * 16;    // 3072   (mlp: K=96 pad, N=32)
    const int TOT = N_W1 + N_RW + N_OW + N_MW + 32 + 32 + 64;
    for (int t = blockIdx.x * blockDim.x + threadIdx.x; t < TOT;
         t += gridDim.x * blockDim.x) {
        int u = t;
        if (u < N_W1) {
            int e = u & 15, lane = (u >> 4) & 31, cj = u >> 9;
            int j = cj & 3, chunk = cj >> 2;
            int k = chunk * 32 + ((lane >> 4) << 4) + e;   // 0..1151
            int n = j * 16 + (lane & 15);                  // 0..63
            int tap = k >> 7, cin = k & 127;
            int dy = tap / 3, dx = tap % 3;
            float sc = bn2_g[n] * rsqrtf(bn2_v[n] + EPS_BN);
            w1p[u] = (_Float16)(out_w1[((n * 128 + cin) * 3 + dy) * 3 + dx] * sc);
            continue;
        }
        u -= N_W1;
        if (u < N_RW) {
            int e = u & 15, lane = (u >> 4) & 31, cj = u >> 9;
            int j = cj & 1, chunk = cj >> 1;
            int k = chunk * 32 + ((lane >> 4) << 4) + e;   // 0..383
            int n = j * 16 + (lane & 15);                  // 0..31
            float sc = bn1_g[n] * rsqrtf(bn1_v[n] + EPS_BN);
            rwp[u] = (_Float16)(reduce_w[n * 384 + k] * sc);
            continue;
        }
        u -= N_RW;
        if (u < N_OW) {
            int e = u & 15, lane = (u >> 4) & 31, tap = u >> 9;
            int cin = ((lane >> 4) << 4) + e;              // 0..31
            int n = lane & 15;
            float v = 0.f;
            if (n < 8) {
                int dy = tap / 3, dx = tap % 3;
                v = off_w[((n * 32 + cin) * 3 + dy) * 3 + dx];
            }
            owp[u] = (_Float16)v;
            continue;
        }
        u -= N_OW;
        if (u < N_MW) {
            int e = u & 15, lane = (u >> 4) & 31, cj = u >> 9;
            int j = cj & 1, chunk = cj >> 1;
            int k = chunk * 32 + ((lane >> 4) << 4) + e;   // 0..95
            int n = j * 16 + (lane & 15);                  // 0..31
            float v = 0.f;
            if (k < 68) {
                float sc = mbn_g[n] * rsqrtf(mbn_v[n] + EPS_BN);
                v = mlp_w[n * 68 + k] * sc;
            }
            mwp[u] = (_Float16)v;
            continue;
        }
        u -= N_MW;
        if (u < 32) {
            float sc = bn1_g[u] * rsqrtf(bn1_v[u] + EPS_BN);
            rbias[u] = (reduce_b[u] - bn1_m[u]) * sc + bn1_b[u];
            continue;
        }
        u -= 32;
        if (u < 32) {
            float sc = mbn_g[u] * rsqrtf(mbn_v[u] + EPS_BN);
            mbias[u] = (mlp_b[u] - mbn_m[u]) * sc + mbn_b[u];
            continue;
        }
        u -= 32;
        {
            float sc = bn2_g[u] * rsqrtf(bn2_v[u] + EPS_BN);
            bias1[u] = (out_b1[u] - bn2_m[u]) * sc + bn2_b[u];
        }
    }
}

// NCHW f32 -> NHWC f16 transposes (output-coalesced)
__global__ void feat_to_f16(const float* __restrict__ feat, _Float16* __restrict__ featT) {
    const long long N = 8LL * 4096 * 384;
    for (long long t = (long long)blockIdx.x * blockDim.x + threadIdx.x; t < N;
         t += (long long)gridDim.x * blockDim.x) {
        int c = (int)(t % 384);
        long long row = t / 384;
        int b = (int)(row >> 12);
        int p = (int)(row & 4095);
        featT[t] = (_Float16)feat[(((long long)(b * 384 + c)) << 12) + p];
    }
}

__global__ void x_to_f16(const float* __restrict__ x, _Float16* __restrict__ x16) {
    const long long N = 8LL * 65536 * 32;
    for (long long t = (long long)blockIdx.x * blockDim.x + threadIdx.x; t < N;
         t += (long long)gridDim.x * blockDim.x) {
        int c = (int)(t & 31);
        long long row = t >> 5;
        int b = (int)(row >> 16);
        int pix = (int)(row & 65535);
        x16[t] = (_Float16)x[(((long long)(b * 32 + c)) << 16) + pix];
    }
}

// ---------------------------------------------------------------------------
// reduce: [32768 x 384] @ [384 x 32] + bias + ReLU -> f16 NHWC [B*4096][32]
// ---------------------------------------------------------------------------
__global__ __launch_bounds__(256) void reduce_gemm(
    const _Float16* __restrict__ featT, const _Float16* __restrict__ rwp,
    const float* __restrict__ rbias, _Float16* __restrict__ f16o)
{
    int wave = blockIdx.x * 8 + (threadIdx.x >> 5);
    int lane = threadIdx.x & 31;
    int m = lane & 15, hi = lane >> 4;
    long long rowbase = (long long)wave * 16;
    const _Float16* arow = featT + (rowbase + m) * 384;

    v8f acc[2];
    acc[0] = zero_acc(); acc[1] = zero_acc();
#pragma unroll
    for (int c = 0; c < 12; ++c) {
        v16h af = load_a_frag(arow + c * 32, hi);
        v16h b0 = load_b_frag(rwp + ((c * 2 + 0) * 32 + lane) * 16);
        v16h b1 = load_b_frag(rwp + ((c * 2 + 1) * 32 + lane) * 16);
        acc[0] = wmma_f16(af, b0, acc[0]);
        acc[1] = wmma_f16(af, b1, acc[1]);
    }
#pragma unroll
    for (int j = 0; j < 2; ++j) {
        int n = j * 16 + (lane & 15);
        float bn = rbias[n];
#pragma unroll
        for (int rr = 0; rr < 8; ++rr) {
            int mm = rr + 8 * hi;
            float v = fmaxf(acc[j][rr] + bn, 0.f);
            f16o[(rowbase + mm) * 32 + n] = (_Float16)v;
        }
    }
}

// ---------------------------------------------------------------------------
// offset conv 3x3 (im2col, K=288) + 2*sigmoid-1 -> ost f32 [B*q][8]
// ---------------------------------------------------------------------------
__global__ __launch_bounds__(256) void offset_gemm(
    const _Float16* __restrict__ x16, const _Float16* __restrict__ owp,
    const float* __restrict__ off_b, float* __restrict__ ost)
{
    int wave = blockIdx.x * 8 + (threadIdx.x >> 5);
    int lane = threadIdx.x & 31;
    int m = lane & 15, hi = lane >> 4;
    long long rowbase = (long long)wave * 16;
    int b = (int)(rowbase >> 16);
    int pixbase = (int)(rowbase & 65535);
    int y = pixbase >> 8, x0 = pixbase & 255;

    v8f acc = zero_acc();
#pragma unroll
    for (int tap = 0; tap < 9; ++tap) {
        int dy = tap / 3 - 1, dx = tap % 3 - 1;
        int yy = y + dy, xx = x0 + m + dx;
        bool inb = ((unsigned)yy < 256u) && ((unsigned)xx < 256u);
        v16h af = zero_frag();
        if (inb)
            af = load_a_frag(x16 + (((long long)b << 16) + yy * 256 + xx) * 32, hi);
        v16h bf = load_b_frag(owp + (tap * 32 + lane) * 16);
        acc = wmma_f16(af, bf, acc);
    }
    int n = lane & 15;
    if (n < 8) {
        float bb = off_b[n];
#pragma unroll
        for (int rr = 0; rr < 8; ++rr) {
            int mm = rr + 8 * hi;
            float v = acc[rr] + bb;
            float o = 2.f / (1.f + __expf(-v)) - 1.f;   // sigmoid*2 - 1
            ost[(rowbase + mm) * 8 + n] = o;
        }
    }
}

// ---------------------------------------------------------------------------
// 4x per-pixel MLP: assemble [16 x 96] rows in LDS (q_feat|x|rel|pad),
// K=96 (3 chunks) x N=32 (2 tiles) WMMA, bias+ReLU -> p16 NHWC [B*q][128]
// ---------------------------------------------------------------------------
__global__ __launch_bounds__(256) void mlp_gemm(
    const float* __restrict__ coord, const float* __restrict__ cell,
    const float* __restrict__ ost, const _Float16* __restrict__ f16f,
    const _Float16* __restrict__ x16, const _Float16* __restrict__ mwp,
    const float* __restrict__ mbias, _Float16* __restrict__ p16)
{
    __shared__ __align__(16) _Float16 lds[8][16][96];
    int w = threadIdx.x >> 5;
    int lane = threadIdx.x & 31;
    int r = lane & 15, hi = lane >> 4;
    long long wave = (long long)blockIdx.x * 8 + w;
    long long rowbase = wave * 16;
    long long grow = rowbase + r;
    int b = (int)(grow >> 16);
    int pix = (int)(grow & 65535);

    float ch = coord[pix * 2 + 0], cw = coord[pix * 2 + 1];
    float cellh = cell[pix * 2 + 0] * 64.f, cellw = cell[pix * 2 + 1] * 64.f;
    const _Float16* xr = x16 + grow * 32;

    const float vx_[4] = {-1.f, -1.f, 1.f, 1.f};
    const float vy_[4] = {-1.f, 1.f, -1.f, 1.f};

    for (int oi = 0; oi < 4; ++oi) {
        // ---- scalar per-row math (each half-wave redundantly computes row r)
        float c0 = fminf(fmaxf(ch + vx_[oi] * (1.f / 64.f) + 1e-6f, -1.f + 1e-6f), 1.f - 1e-6f);
        float c1 = fminf(fmaxf(cw + vy_[oi] * (1.f / 64.f) + 1e-6f, -1.f + 1e-6f), 1.f - 1e-6f);
        float ox = ost[grow * 8 + oi * 2 + 0];
        float oy = ost[grow * 8 + oi * 2 + 1];
        float gx = c1 + ox, gy = c0 + oy;
        float ixf = rintf((gx + 1.f) * 32.f - 0.5f);
        float iyf = rintf((gy + 1.f) * 32.f - 0.5f);
        bool valid = (ixf >= 0.f) && (ixf < 64.f) && (iyf >= 0.f) && (iyf < 64.f);
        int ix = min(max((int)ixf, 0), 63);
        int iy = min(max((int)iyf, 0), 63);
        const _Float16* fr = f16f + (((long long)b << 12) + iy * 64 + ix) * 32;

        // ---- stage row into LDS
        float4* dq = (float4*)&lds[w][r][hi * 16];
        if (valid) {
            dq[0] = *(const float4*)(fr + hi * 16);
            dq[1] = *(const float4*)(fr + hi * 16 + 8);
        } else {
            dq[0] = make_float4(0.f, 0.f, 0.f, 0.f);
            dq[1] = make_float4(0.f, 0.f, 0.f, 0.f);
        }
        float4* dx4 = (float4*)&lds[w][r][32 + hi * 16];
        dx4[0] = *(const float4*)(xr + hi * 16);
        dx4[1] = *(const float4*)(xr + hi * 16 + 8);
        if (hi == 0) {
            float ix2 = fminf(fmaxf(rintf((c1 + 1.f) * 32.f - 0.5f), 0.f), 63.f);
            float iy2 = fminf(fmaxf(rintf((c0 + 1.f) * 32.f - 0.5f), 0.f), 63.f);
            float qch = -1.f + (2.f * iy2 + 1.f) / 64.f;
            float qcw = -1.f + (2.f * ix2 + 1.f) / 64.f;
            lds[w][r][64] = (_Float16)((ch - qch) * 64.f);
            lds[w][r][65] = (_Float16)((cw - qcw) * 64.f);
            lds[w][r][66] = (_Float16)cellh;
            lds[w][r][67] = (_Float16)cellw;
            for (int z = 68; z < 80; ++z) lds[w][r][z] = (_Float16)0.f;
        } else {
            float4* dz = (float4*)&lds[w][r][80];
            dz[0] = make_float4(0.f, 0.f, 0.f, 0.f);
            dz[1] = make_float4(0.f, 0.f, 0.f, 0.f);
        }
        __syncthreads();

        // ---- WMMA GEMM [16 x 96] @ [96 x 32]
        v8f acc[2];
        acc[0] = zero_acc(); acc[1] = zero_acc();
        const _Float16* arow = &lds[w][r][0];
#pragma unroll
        for (int c = 0; c < 3; ++c) {
            v16h af = load_a_frag(arow + c * 32, hi);
            v16h b0 = load_b_frag(mwp + ((c * 2 + 0) * 32 + lane) * 16);
            v16h b1 = load_b_frag(mwp + ((c * 2 + 1) * 32 + lane) * 16);
            acc[0] = wmma_f16(af, b0, acc[0]);
            acc[1] = wmma_f16(af, b1, acc[1]);
        }
#pragma unroll
        for (int j = 0; j < 2; ++j) {
            int n = j * 16 + (lane & 15);
            float bn = mbias[n];
#pragma unroll
            for (int rr = 0; rr < 8; ++rr) {
                int mm = rr + 8 * hi;
                float v = fmaxf(acc[j][rr] + bn, 0.f);
                p16[(rowbase + mm) * 128 + oi * 32 + n] = (_Float16)v;
            }
        }
        __syncthreads();
    }
}

// ---------------------------------------------------------------------------
// out conv 3x3 (K=1152, N=64) + bias + ReLU fused with final 1x1 (dot w2) -> f32
// ---------------------------------------------------------------------------
__global__ __launch_bounds__(256) void outconv_gemm(
    const _Float16* __restrict__ p16, const _Float16* __restrict__ w1p,
    const float* __restrict__ bias1, const float* __restrict__ w2,
    const float* __restrict__ b2, float* __restrict__ out)
{
    int wave = blockIdx.x * 8 + (threadIdx.x >> 5);
    int lane = threadIdx.x & 31;
    int m = lane & 15, hi = lane >> 4;
    long long rowbase = (long long)wave * 16;
    int b = (int)(rowbase >> 16);
    int pixbase = (int)(rowbase & 65535);
    int y = pixbase >> 8, x0 = pixbase & 255;

    v8f acc[4];
#pragma unroll
    for (int j = 0; j < 4; ++j) acc[j] = zero_acc();

#pragma unroll
    for (int tap = 0; tap < 9; ++tap) {
        int dy = tap / 3 - 1, dx = tap % 3 - 1;
        int yy = y + dy, xx = x0 + m + dx;
        bool inb = ((unsigned)yy < 256u) && ((unsigned)xx < 256u);
        const _Float16* prow = p16 + (((long long)b << 16) + yy * 256 + xx) * 128;
#pragma unroll
        for (int g = 0; g < 4; ++g) {
            v16h af = zero_frag();
            if (inb) af = load_a_frag(prow + g * 32, hi);
            int chunk = tap * 4 + g;
#pragma unroll
            for (int j = 0; j < 4; ++j) {
                v16h bf = load_b_frag(w1p + ((chunk * 4 + j) * 32 + lane) * 16);
                acc[j] = wmma_f16(af, bf, acc[j]);
            }
        }
    }

    float bb = b2[0];
    float bs[4], ws_[4];
#pragma unroll
    for (int j = 0; j < 4; ++j) {
        int n = j * 16 + (lane & 15);
        bs[j] = bias1[n];
        ws_[j] = w2[n];
    }
#pragma unroll
    for (int rr = 0; rr < 8; ++rr) {
        float s = 0.f;
#pragma unroll
        for (int j = 0; j < 4; ++j)
            s += fmaxf(acc[j][rr] + bs[j], 0.f) * ws_[j];
        // reduce over the 16 lanes sharing `hi` (n index space)
#pragma unroll
        for (int off = 1; off < 16; off <<= 1) s += __shfl_xor(s, off, 32);
        if ((lane & 15) == 0) {
            int mm = rr + 8 * hi;
            out[((long long)b << 16) + y * 256 + x0 + mm] = s + bb;
        }
    }
}

// ---------------------------------------------------------------------------
extern "C" void kernel_launch(void* const* d_in, const int* in_sizes, int n_in,
                              void* d_out, int out_size, void* d_ws, size_t ws_size,
                              hipStream_t stream) {
    (void)in_sizes; (void)n_in; (void)out_size; (void)ws_size;
    const float* feat     = (const float*)d_in[0];
    const float* x        = (const float*)d_in[1];
    const float* coord    = (const float*)d_in[2];
    const float* cell     = (const float*)d_in[3];
    const float* reduce_w = (const float*)d_in[4];
    const float* reduce_b = (const float*)d_in[5];
    const float* bn1_g    = (const float*)d_in[6];
    const float* bn1_b    = (const float*)d_in[7];
    const float* bn1_m    = (const float*)d_in[8];
    const float* bn1_v    = (const float*)d_in[9];
    const float* off_b    = (const float*)d_in[11];
    const float* off_w    = (const float*)d_in[10];
    const float* mlp_w    = (const float*)d_in[12];
    const float* mlp_b    = (const float*)d_in[13];
    const float* mbn_g    = (const float*)d_in[14];
    const float* mbn_b    = (const float*)d_in[15];
    const float* mbn_m    = (const float*)d_in[16];
    const float* mbn_v    = (const float*)d_in[17];
    const float* out_w1   = (const float*)d_in[18];
    const float* out_b1   = (const float*)d_in[19];
    const float* bn2_g    = (const float*)d_in[20];
    const float* bn2_b    = (const float*)d_in[21];
    const float* bn2_m    = (const float*)d_in[22];
    const float* bn2_v    = (const float*)d_in[23];
    const float* out_w2   = (const float*)d_in[24];
    const float* out_b2   = (const float*)d_in[25];

    char* ws = (char*)d_ws;
    _Float16* featT = (_Float16*)(ws + 0);            //  25,165,824 B
    _Float16* x16   = (_Float16*)(ws + 25165824);     //  33,554,432 B
    _Float16* f16f  = (_Float16*)(ws + 58720256);     //   2,097,152 B
    float*    ost   = (float*)   (ws + 60817408);     //  16,777,216 B
    _Float16* p16   = (_Float16*)(ws + 77594624);     // 134,217,728 B
    _Float16* w1p   = (_Float16*)(ws + 211812352);    //     147,456 B
    _Float16* rwp   = (_Float16*)(ws + 211959808);    //      24,576 B
    _Float16* owp   = (_Float16*)(ws + 211984384);    //       9,216 B
    _Float16* mwp   = (_Float16*)(ws + 211993600);    //       6,144 B
    float*    rbias = (float*)   (ws + 211999744);
    float*    mbias = (float*)   (ws + 211999872);
    float*    bias1 = (float*)   (ws + 212000000);

    pack_weights<<<128, 256, 0, stream>>>(
        reduce_w, reduce_b, bn1_g, bn1_b, bn1_m, bn1_v, off_w,
        mlp_w, mlp_b, mbn_g, mbn_b, mbn_m, mbn_v,
        out_w1, out_b1, bn2_g, bn2_b, bn2_m, bn2_v,
        w1p, rwp, owp, mwp, rbias, mbias, bias1);
    feat_to_f16<<<4096, 256, 0, stream>>>(feat, featT);
    x_to_f16<<<8192, 256, 0, stream>>>(x, x16);
    reduce_gemm<<<256, 256, 0, stream>>>(featT, rwp, rbias, f16f);
    offset_gemm<<<4096, 256, 0, stream>>>(x16, owp, off_b, ost);
    mlp_gemm<<<4096, 256, 0, stream>>>(coord, cell, ost, f16f, x16, mwp, mbias, p16);
    outconv_gemm<<<4096, 256, 0, stream>>>(p16, w1p, bias1, out_w2, out_b2,
                                           (float*)d_out);
}
